// InvariantPointAttention_8529805050430
// MI455X (gfx1250) — matrix operand order repaired and checked by reading, not compiled, not run on hardware
//
#include <hip/hip_runtime.h>
#include <hip/hip_bf16.h>
#include <math.h>

typedef __attribute__((ext_vector_type(16))) _Float16 v16h;
typedef __attribute__((ext_vector_type(8)))  float    v8f;

#define N_RES 512
#define CS    384
#define CZd   128
#define CIPA  16
#define PQd   4
#define PVd   8
#define NH    12

// ---------------------------------------------------------------------------
// Generic batched WMMA GEMM (fp32 in/out, f16 multiply, f32 accumulate).
// C[b][m][n] = alpha*alphaB[b] * sum_k A[b](m,k)*B[b](k,n) (+ beta*C) (+ bias[n])
// A addressed as Ab[m*rsA + k*csA], B as Bb[k*rsB + n*csB].
//
// One wave per 32x32 macro-tile: 2 A-fragments x 2 B-fragments -> 4
// v_wmma_f32_16x16x32_f16 per 32-K step (2x operand reuse vs 16x16 tiles).
//
// Fragment layouts (CDNA5 ISA 7.12.2, wave32):
//   A (16x32 f16): lane L: m=L&15, kb=(L>=16)?8:0, half e: k=kb+(e<8?e:e+8)
//   B (32x16 f16): lane L: n=L&15, kb=(L>=16)?8:0, half e: k=kb+(e<8?e:e+8)
//   C/D (16x16 f32): lane L: n=L&15, vgpr r: m=(L<16)?r:r+8
//
// Out-of-range rows/cols: load pointers are CLAMPED (memory-safe) and the
// garbage results are simply never written (row m of A only affects row m of
// D; col n of B only affects col n of D) -> no per-element masking in the
// main loop. Only the K dimension (which is reduced) is zero-masked, and only
// in the K%32 tail. ACONT (csA==1) / BCONT (rsB==1) use float4 loads; all
// call sites guarantee 16B alignment of the k-runs.
// ---------------------------------------------------------------------------

template <bool CONT>
__device__ __forceinline__ void load_frag_a(const float* __restrict__ p,
                                            int k0, int kbA, int csA,
                                            float (&av)[16])
{
  if (CONT) {
    const float4 a0 = *(const float4*)(p + k0 + kbA + 0);
    const float4 a1 = *(const float4*)(p + k0 + kbA + 4);
    const float4 a2 = *(const float4*)(p + k0 + kbA + 16);
    const float4 a3 = *(const float4*)(p + k0 + kbA + 20);
    av[0]=a0.x; av[1]=a0.y; av[2]=a0.z; av[3]=a0.w;
    av[4]=a1.x; av[5]=a1.y; av[6]=a1.z; av[7]=a1.w;
    av[8]=a2.x; av[9]=a2.y; av[10]=a2.z; av[11]=a2.w;
    av[12]=a3.x; av[13]=a3.y; av[14]=a3.z; av[15]=a3.w;
  } else {
#pragma unroll
    for (int e = 0; e < 16; ++e)
      av[e] = p[(k0 + kbA + ((e < 8) ? e : (e + 8))) * csA];
  }
}

template <bool CONT>
__device__ __forceinline__ void load_frag_b(const float* __restrict__ p,
                                            int k0, int kbB, int rsB,
                                            float (&bv)[16])
{
  if (CONT) {
    const float4 b0 = *(const float4*)(p + k0 + kbB + 0);
    const float4 b1 = *(const float4*)(p + k0 + kbB + 4);
    const float4 b2 = *(const float4*)(p + k0 + kbB + 16);
    const float4 b3 = *(const float4*)(p + k0 + kbB + 20);
    bv[0]=b0.x; bv[1]=b0.y; bv[2]=b0.z; bv[3]=b0.w;
    bv[4]=b1.x; bv[5]=b1.y; bv[6]=b1.z; bv[7]=b1.w;
    bv[8]=b2.x; bv[9]=b2.y; bv[10]=b2.z; bv[11]=b2.w;
    bv[12]=b3.x; bv[13]=b3.y; bv[14]=b3.z; bv[15]=b3.w;
  } else {
#pragma unroll
    for (int e = 0; e < 16; ++e)
      bv[e] = p[(k0 + kbB + ((e < 8) ? e : (e + 8))) * rsB];
  }
}

__device__ __forceinline__ v16h cvt_frag(const float (&x)[16])
{
  v16h f;
#pragma unroll
  for (int e = 0; e < 16; ++e) f[e] = (_Float16)x[e];
  return f;
}

template <bool ACONT, bool BCONT>
__global__ void ipa_wmma_gemm(
    const float* __restrict__ A, long long sAb, int rsA, int csA,
    const float* __restrict__ B, long long sBb, int rsB, int csB,
    float* __restrict__ C, long long sCb, int ldc,
    int M, int Nc, int K,
    float alpha, const float* __restrict__ alphaB,
    int betaFlag, const float* __restrict__ bias)
{
  const int lane = threadIdx.x;  // 0..31
  const int b    = blockIdx.z;
  const float* Ab = A + (long long)b * sAb;
  const float* Bb = B + (long long)b * sBb;
  float*       Cb = C + (long long)b * sCb;

  const int m0 = blockIdx.y * 32;
  const int n0 = blockIdx.x * 32;
  const int l15 = lane & 15;
  const int kbA = (lane >= 16) ? 8 : 0;
  const int kbB = (lane >= 16) ? 8 : 0;

  // clamped per-lane base pointers (always legal to read)
  const int am0 = m0 + l15,      am1 = m0 + 16 + l15;
  const int bn0 = n0 + l15,      bn1 = n0 + 16 + l15;
  const float* aRow0 = Ab + (long long)((am0 < M) ? am0 : (M - 1)) * rsA;
  const float* aRow1 = Ab + (long long)((am1 < M) ? am1 : (M - 1)) * rsA;
  const float* bCol0 = Bb + (long long)((bn0 < Nc) ? bn0 : (Nc - 1)) * csB;
  const float* bCol1 = Bb + (long long)((bn1 < Nc) ? bn1 : (Nc - 1)) * csB;

  v8f acc00 = {0,0,0,0,0,0,0,0}, acc01 = acc00, acc10 = acc00, acc11 = acc00;
  const int Kfull = K & ~31;

  for (int k0 = 0; k0 < Kfull; k0 += 32) {
    float av0[16], av1[16], bv0[16], bv1[16];
    load_frag_a<ACONT>(aRow0, k0, kbA, csA, av0);
    load_frag_a<ACONT>(aRow1, k0, kbA, csA, av1);
    load_frag_b<BCONT>(bCol0, k0, kbB, rsB, bv0);
    load_frag_b<BCONT>(bCol1, k0, kbB, rsB, bv1);
    if (!BCONT) {  // speculative stream-prefetch of next B K-slab (pair_rep)
      __builtin_prefetch(bCol0 + (k0 + 32 + kbB) * rsB, 0, 1);
      __builtin_prefetch(bCol1 + (k0 + 32 + kbB) * rsB, 0, 1);
    }
    const v16h af0 = cvt_frag(av0), af1 = cvt_frag(av1);
    const v16h bf0 = cvt_frag(bv0), bf1 = cvt_frag(bv1);
    acc00 = __builtin_amdgcn_wmma_f32_16x16x32_f16(false, af0, false, bf0, (short)0, acc00, false, false);
    acc01 = __builtin_amdgcn_wmma_f32_16x16x32_f16(false, af0, false, bf1, (short)0, acc01, false, false);
    acc10 = __builtin_amdgcn_wmma_f32_16x16x32_f16(false, af1, false, bf0, (short)0, acc10, false, false);
    acc11 = __builtin_amdgcn_wmma_f32_16x16x32_f16(false, af1, false, bf1, (short)0, acc11, false, false);
  }

  if (Kfull < K) {  // K-tail: clamped gathers, zero-masked along K only
    float av0[16], av1[16], bv0[16], bv1[16];
    v16h af0, af1, bf0, bf1;
#pragma unroll
    for (int e = 0; e < 16; ++e) {
      const int ka  = Kfull + kbA + ((e < 8) ? e : (e + 8));
      const int kaC = (ka < K) ? ka : (K - 1);
      av0[e] = aRow0[kaC * csA];
      av1[e] = aRow1[kaC * csA];
      const int kb  = Kfull + kbB + ((e < 8) ? e : (e + 8));
      const int kbC = (kb < K) ? kb : (K - 1);
      bv0[e] = bCol0[kbC * rsB];
      bv1[e] = bCol1[kbC * rsB];
      af0[e] = (_Float16)((ka < K) ? av0[e] : 0.0f);
      af1[e] = (_Float16)((ka < K) ? av1[e] : 0.0f);
      bf0[e] = (_Float16)((kb < K) ? bv0[e] : 0.0f);
      bf1[e] = (_Float16)((kb < K) ? bv1[e] : 0.0f);
    }
    acc00 = __builtin_amdgcn_wmma_f32_16x16x32_f16(false, af0, false, bf0, (short)0, acc00, false, false);
    acc01 = __builtin_amdgcn_wmma_f32_16x16x32_f16(false, af0, false, bf1, (short)0, acc01, false, false);
    acc10 = __builtin_amdgcn_wmma_f32_16x16x32_f16(false, af1, false, bf0, (short)0, acc10, false, false);
    acc11 = __builtin_amdgcn_wmma_f32_16x16x32_f16(false, af1, false, bf1, (short)0, acc11, false, false);
  }

  // ---- epilogue: 4 sub-tiles ----
  const float al = alpha * (alphaB ? alphaB[b] : 1.0f);
  const v8f* accs[4] = {&acc00, &acc01, &acc10, &acc11};
#pragma unroll
  for (int mt = 0; mt < 2; ++mt) {
#pragma unroll
    for (int nt = 0; nt < 2; ++nt) {
      const v8f& acc = *accs[mt * 2 + nt];
      const int bn = n0 + nt * 16 + l15;
      if (bn >= Nc) continue;
      const float bias_n = bias ? bias[bn] : 0.0f;
      const int mbase = m0 + mt * 16 + ((lane < 16) ? 0 : 8);
      float val[8];
#pragma unroll
      for (int r = 0; r < 8; ++r) val[r] = al * acc[r] + bias_n;
      if (betaFlag) {
#pragma unroll
        for (int r = 0; r < 8; ++r) {
          const int cm = mbase + r;
          if (cm < M) val[r] += Cb[cm * ldc + bn];
        }
      }
#pragma unroll
      for (int r = 0; r < 8; ++r) {
        const int cm = mbase + r;
        if (cm < M) Cb[cm * ldc + bn] = val[r];
      }
    }
  }
}

// ---------------------------------------------------------------------------
// Elementwise kernels
// ---------------------------------------------------------------------------

// out[n][hp][i] = R_n[i][j] * x[n][hp][j] + t_n[i]   (hp = h*P+p, W = H*P*3)
__global__ void ipa_apply_frames(const float* __restrict__ x,
                                 const float* __restrict__ rot,
                                 const float* __restrict__ trans,
                                 float* __restrict__ out, int P)
{
  const int W = NH * P * 3;
  int idx = blockIdx.x * blockDim.x + threadIdx.x;
  int total = N_RES * NH * P;
  if (idx >= total) return;
  int n  = idx / (NH * P);
  int hp = idx % (NH * P);
  const float* R = rot + n * 9;
  const float* t = trans + n * 3;
  const float* xi = x + (long long)n * W + hp * 3;
  float x0 = xi[0], x1 = xi[1], x2 = xi[2];
  float* o = out + (long long)n * W + hp * 3;
#pragma unroll
  for (int i = 0; i < 3; ++i)
    o[i] = R[i * 3 + 0] * x0 + R[i * 3 + 1] * x1 + R[i * 3 + 2] * x2 + t[i];
}

// sq[n*H+h] = sum_{j<12} L[n][h*12+j]^2   (L is lq or lk, width 144)
__global__ void ipa_sqnorm(const float* __restrict__ L, float* __restrict__ sq)
{
  int idx = blockIdx.x * blockDim.x + threadIdx.x;
  if (idx >= N_RES * NH) return;
  int n = idx / NH, h = idx % NH;
  const float* p = L + (long long)n * (NH * PQd * 3) + h * (PQd * 3);
  float s = 0.f;
#pragma unroll
  for (int j = 0; j < PQd * 3; ++j) s += p[j] * p[j];
  sq[idx] = s;
}

__global__ void ipa_softplus(const float* __restrict__ x, float* __restrict__ sp)
{
  int h = threadIdx.x;
  if (h < NH) {
    float v = x[h];
    sp[h] = (v > 20.f) ? v : log1pf(expf(v));
  }
}

// Fused: logits[n][h][m] += SF*sp[h]*(sqq[n,h]+sqk[m,h]); then softmax over m.
// One wave per row; 4 waves per block. N=512 -> 16 elements per lane.
__global__ void ipa_softmax(float* __restrict__ logits,
                            const float* __restrict__ sqq,
                            const float* __restrict__ sqk,
                            const float* __restrict__ sp,
                            float scale_frame)
{
  const int wv = threadIdx.x >> 5;
  const int lane = threadIdx.x & 31;
  const int row = blockIdx.x * 4 + wv;
  if (row >= N_RES * NH) return;
  const int h = row % NH;
  float* L = logits + (long long)row * N_RES;
  const float sf = scale_frame * sp[h];
  const float qq = sqq[row];

  float vals[16];
  float mx = -1e30f;
#pragma unroll
  for (int i = 0; i < 16; ++i) {
    const int m = lane + 32 * i;
    float v = L[m] + sf * (qq + sqk[m * NH + h]);
    vals[i] = v;
    mx = fmaxf(mx, v);
  }
#pragma unroll
  for (int off = 16; off > 0; off >>= 1)
    mx = fmaxf(mx, __shfl_xor(mx, off, 32));
  float sum = 0.f;
#pragma unroll
  for (int i = 0; i < 16; ++i) { vals[i] = expf(vals[i] - mx); sum += vals[i]; }
#pragma unroll
  for (int off = 16; off > 0; off >>= 1)
    sum += __shfl_xor(sum, off, 32);
  const float inv = 1.0f / sum;
#pragma unroll
  for (int i = 0; i < 16; ++i) L[lane + 32 * i] = vals[i] * inv;
}

// o_frame = R^T (o_local - t); write o_frame + o_norm into att.
// att row layout per (n,h): [0:16)=o_single, [16:144)=o_pair,
// [144:168)=o_frame, [168:176)=o_norm; row width = 12*176 = 2112.
__global__ void ipa_finalize(const float* __restrict__ olb,
                             const float* __restrict__ rot,
                             const float* __restrict__ trans,
                             float* __restrict__ att)
{
  int idx = blockIdx.x * blockDim.x + threadIdx.x;
  if (idx >= N_RES * NH * PVd) return;
  int n = idx / (NH * PVd);
  int hp = idx % (NH * PVd);
  int h = hp / PVd, p = hp % PVd;
  const float* R = rot + n * 9;
  const float* t = trans + n * 3;
  const float* ol = olb + (long long)n * (NH * PVd * 3) + h * (PVd * 3) + p * 3;
  float d0 = ol[0] - t[0], d1 = ol[1] - t[1], d2 = ol[2] - t[2];
  float* arow = att + (long long)n * 2112 + h * 176;
  float nrm2 = 0.f;
#pragma unroll
  for (int i = 0; i < 3; ++i) {
    float of = R[0 * 3 + i] * d0 + R[1 * 3 + i] * d1 + R[2 * 3 + i] * d2;
    arow[144 + p * 3 + i] = of;
    nrm2 += of * of;
  }
  arow[168 + p] = sqrtf(nrm2);
}

// ---------------------------------------------------------------------------
// Host-side orchestration
// ---------------------------------------------------------------------------
static inline void launch_gemm(hipStream_t s,
    const float* A, long long sAb, int rsA, int csA,
    const float* B, long long sBb, int rsB, int csB,
    float* C, long long sCb, int ldc,
    int M, int Nc, int K, int batch,
    float alpha, const float* alphaB, int betaFlag, const float* bias)
{
  dim3 grid((Nc + 31) / 32, (M + 31) / 32, batch);
  dim3 blk(32);
  const bool ac = (csA == 1), bc = (rsB == 1);
  if (ac && bc)
    ipa_wmma_gemm<true, true><<<grid, blk, 0, s>>>(A, sAb, rsA, csA, B, sBb, rsB, csB,
        C, sCb, ldc, M, Nc, K, alpha, alphaB, betaFlag, bias);
  else if (ac)
    ipa_wmma_gemm<true, false><<<grid, blk, 0, s>>>(A, sAb, rsA, csA, B, sBb, rsB, csB,
        C, sCb, ldc, M, Nc, K, alpha, alphaB, betaFlag, bias);
  else if (bc)
    ipa_wmma_gemm<false, true><<<grid, blk, 0, s>>>(A, sAb, rsA, csA, B, sBb, rsB, csB,
        C, sCb, ldc, M, Nc, K, alpha, alphaB, betaFlag, bias);
  else
    ipa_wmma_gemm<false, false><<<grid, blk, 0, s>>>(A, sAb, rsA, csA, B, sBb, rsB, csB,
        C, sCb, ldc, M, Nc, K, alpha, alphaB, betaFlag, bias);
}

extern "C" void kernel_launch(void* const* d_in, const int* in_sizes, int n_in,
                              void* d_out, int out_size, void* d_ws, size_t ws_size,
                              hipStream_t stream) {
  (void)in_sizes; (void)n_in; (void)out_size; (void)ws_size;
  const float* single = (const float*)d_in[0];   // [512,384]
  const float* pair   = (const float*)d_in[1];   // [512,512,128]
  const float* rot    = (const float*)d_in[2];   // [512,3,3]
  const float* trans  = (const float*)d_in[3];   // [512,3]
  const float* Wq     = (const float*)d_in[4];   // [384,192]
  const float* Wk     = (const float*)d_in[5];
  const float* Wv     = (const float*)d_in[6];
  const float* Wqp    = (const float*)d_in[7];   // [384,144]
  const float* Wkp    = (const float*)d_in[8];
  const float* Wvp    = (const float*)d_in[9];   // [384,288]
  const float* Wb     = (const float*)d_in[10];  // [128,12]
  const float* Wout   = (const float*)d_in[11];  // [2112,384]
  const float* bout   = (const float*)d_in[12];  // [384]
  const float* shead  = (const float*)d_in[13];  // [12]
  float* out = (float*)d_out;                    // [512,384]

  float* ws = (float*)d_ws;
  float* q      = ws;            // 512*192
  float* k      = q   + 98304;   // 512*192
  float* v      = k   + 98304;   // 512*192
  float* qp     = v   + 98304;   // 512*144
  float* kp     = qp  + 73728;   // 512*144
  float* vp     = kp  + 73728;   // 512*288
  float* lq     = vp  + 147456;  // 512*144
  float* lk     = lq  + 73728;   // 512*144
  float* lvp    = lk  + 73728;   // 512*288
  float* sqq    = lvp + 147456;  // 6144
  float* sqk    = sqq + 6144;    // 6144
  float* sp     = sqk + 6144;    // 16
  float* olb    = sp  + 16;      // 512*288
  float* att    = olb + 147456;  // 512*2112
  float* logits = att + 1081344; // 512*12*512

  const float SCALE_SINGLE = 0.25f;               // 1/sqrt(16)
  const float SCALE_FRAME  = -0.117851130197758f; // -1/sqrt(72)

  // ---- projections: X[512x384] @ W[384xKc] ----
  launch_gemm(stream, single, 0, CS, 1,  Wq,  0, 192, 1, q,  0, 192, N_RES, 192, CS, 1, 1.f, nullptr, 0, nullptr);
  launch_gemm(stream, single, 0, CS, 1,  Wk,  0, 192, 1, k,  0, 192, N_RES, 192, CS, 1, 1.f, nullptr, 0, nullptr);
  launch_gemm(stream, single, 0, CS, 1,  Wv,  0, 192, 1, v,  0, 192, N_RES, 192, CS, 1, 1.f, nullptr, 0, nullptr);
  launch_gemm(stream, single, 0, CS, 1,  Wqp, 0, 144, 1, qp, 0, 144, N_RES, 144, CS, 1, 1.f, nullptr, 0, nullptr);
  launch_gemm(stream, single, 0, CS, 1,  Wkp, 0, 144, 1, kp, 0, 144, N_RES, 144, CS, 1, 1.f, nullptr, 0, nullptr);
  launch_gemm(stream, single, 0, CS, 1,  Wvp, 0, 288, 1, vp, 0, 288, N_RES, 288, CS, 1, 1.f, nullptr, 0, nullptr);

  // ---- apply frames ----
  {
    int tq = N_RES * NH * PQd, tv = N_RES * NH * PVd;
    ipa_apply_frames<<<(tq + 255) / 256, 256, 0, stream>>>(qp, rot, trans, lq, PQd);
    ipa_apply_frames<<<(tq + 255) / 256, 256, 0, stream>>>(kp, rot, trans, lk, PQd);
    ipa_apply_frames<<<(tv + 255) / 256, 256, 0, stream>>>(vp, rot, trans, lvp, PVd);
    ipa_sqnorm<<<(N_RES * NH + 255) / 256, 256, 0, stream>>>(lq, sqq);
    ipa_sqnorm<<<(N_RES * NH + 255) / 256, 256, 0, stream>>>(lk, sqk);
    ipa_softplus<<<1, 32, 0, stream>>>(shead, sp);
  }

  // ---- logits = w_single (beta=0), batched over heads; K=16 zero-padded ----
  launch_gemm(stream, q, 16, 192, 1, k, 16, 1, 192,
              logits, 512, NH * N_RES, N_RES, N_RES, CIPA, NH,
              SCALE_SINGLE, nullptr, 0, nullptr);

  // ---- logits += w_pair: per-n Wb^T[12x128] x pair[n]^T[128x512] ----
  launch_gemm(stream, Wb, 0, 1, NH, pair, (long long)N_RES * CZd, 1, CZd,
              logits, NH * N_RES, N_RES, NH, N_RES, CZd, N_RES,
              1.0f, nullptr, 1, nullptr);

  // ---- logits += (-2*SF*sp_h) * lq . lk  (batched over heads, K=12) ----
  launch_gemm(stream, lq, 12, 144, 1, lk, 12, 1, 144,
              logits, 512, NH * N_RES, N_RES, N_RES, PQd * 3, NH,
              -2.0f * SCALE_FRAME, sp, 1, nullptr);

  // ---- fused |lq|^2+|lk|^2 bias + softmax over m ----
  ipa_softmax<<<(N_RES * NH + 3) / 4, 128, 0, stream>>>(logits, sqq, sqk, sp, SCALE_FRAME);

  // ---- o_single -> att[:, h*176 + 0:16]  (batched over heads, K=512) ----
  launch_gemm(stream, logits, 512, NH * N_RES, 1, v, 16, 192, 1,
              att, 176, 2112, N_RES, CIPA, N_RES, NH,
              1.0f, nullptr, 0, nullptr);

  // ---- o_local -> olb  (batched over heads, 24 cols) ----
  launch_gemm(stream, logits, 512, NH * N_RES, 1, lvp, 24, 288, 1,
              olb, 24, 288, N_RES, PVd * 3, N_RES, NH,
              1.0f, nullptr, 0, nullptr);

  // ---- o_pair -> att[:, h*176 + 16:144]  (batched over n; streams pair) ----
  launch_gemm(stream, logits, NH * N_RES, N_RES, 1,
              pair, (long long)N_RES * CZd, CZd, 1,
              att + 16, 2112, 176, NH, CZd, N_RES, N_RES,
              1.0f, nullptr, 0, nullptr);

  // ---- inverse frames + norms -> att[:, 144:176] ----
  ipa_finalize<<<(N_RES * NH * PVd + 255) / 256, 256, 0, stream>>>(olb, rot, trans, att);

  // ---- out = att[512x2112] @ Wout[2112x384] + bout ----
  launch_gemm(stream, att, 0, 2112, 1, Wout, 0, CS, 1,
              out, 0, CS, N_RES, CS, 2112, 1,
              1.0f, nullptr, 0, bout);
}
